// GNN_15212774162888
// MI455X (gfx1250) — matrix operand order repaired and checked by reading, not compiled, not run on hardware
//
#include <hip/hip_runtime.h>
#include <hip/hip_bf16.h>
#include <math.h>

// GCN (2-layer) for MI455X / gfx1250.
//   layer1: t1 = x @ W1          (bf16 WMMA, f32 accum, B staged in LDS)   [N,64]
//           agg1 = scatter(norm * t1[src]) + self + b1, ReLU  (L2-resident f32 atomics)
//   layer2: t2 = h1 @ W2         (bf16 WMMA)                               [N,40]
//           agg2 = scatter(...) + self + b2, ReLU, log_softmax
//
// Sizes fixed by reference: F=500, H=64, C=40.

#define F_IN   500
#define H_HID  64
#define C_OUT  40

typedef __attribute__((ext_vector_type(16))) __bf16    v16bf;
typedef __attribute__((ext_vector_type(8)))  float     v8f;
typedef __attribute__((ext_vector_type(8)))  unsigned  v8u;

// Pack two f32 into one dword of two bf16.
#if __has_builtin(__builtin_amdgcn_cvt_pk_bf16_f32)
typedef __attribute__((ext_vector_type(2))) __bf16 v2bf;
__device__ __forceinline__ unsigned pk_bf16(float a, float b) {
    v2bf r = __builtin_amdgcn_cvt_pk_bf16_f32(a, b);   // v_cvt_pk_bf16_f32
    return __builtin_bit_cast(unsigned, r);
}
#else
// Round-to-nearest (+0x8000 carry into mantissa) then one v_perm_b32 to pack
// the two high halves: 3 VALU per dword instead of the 6-op RNE sequence.
__device__ __forceinline__ unsigned pk_bf16(float a, float b) {
    unsigned ua = __float_as_uint(a) + 0x8000u;
    unsigned ub = __float_as_uint(b) + 0x8000u;
    return __builtin_amdgcn_perm(ub, ua, 0x07060302u);  // {ub[31:16], ua[31:16]}
}
#endif

// Relaxed agent-scope f32 atomic add -> non-returning global_atomic_add_f32.
__device__ __forceinline__ void fadd_atomic(float* p, float v) {
    __hip_atomic_fetch_add(p, v, __ATOMIC_RELAXED, __HIP_MEMORY_SCOPE_AGENT);
}

// ---------------- degree / norm ----------------
__global__ void k_init_deg(float* deg, int n) {
    int i = blockIdx.x * blockDim.x + threadIdx.x;
    if (i < n) deg[i] = 1.0f;                       // self loop
}
__global__ void k_count_deg(float* deg, const int* __restrict__ dst, int e) {
    int i = blockIdx.x * blockDim.x + threadIdx.x;
    if (i < e) fadd_atomic(&deg[dst[i]], 1.0f);
}
__global__ void k_dinv(float* deg, int n) {
    int i = blockIdx.x * blockDim.x + threadIdx.x;
    if (i < n) deg[i] = rsqrtf(deg[i]);             // deg >= 1 always
}
__global__ void k_zero(float* p, int n) {
    int i = blockIdx.x * blockDim.x + threadIdx.x;
    if (i < n) p[i] = 0.0f;
}

// ---------------- weight pre-pack into WMMA B-operand layout ----------------
// B (bf16, 32x16 per tile), ISA 7.12.5: VGPR j, lane L: n = L&15,
// k = kstep*32 + (j<4?0:16) + (L>>4)*8 + 2*(j&3) (+1 in high half of dword).
// Output layout: [ks][nt][lane][8 dwords]  -> each GEMM lane loads 32B contiguous.
__global__ void k_pack_b(const float* __restrict__ W, unsigned* __restrict__ out,
                         int kvalid, int ncols, int ksteps, int ntiles) {
    int t = blockIdx.x * blockDim.x + threadIdx.x;
    int total = ksteps * ntiles * 32;
    if (t >= total) return;
    int lane = t & 31;
    int nt   = (t >> 5) % ntiles;
    int ks   = (t >> 5) / ntiles;
    int hh   = lane >> 4;
    int n    = nt * 16 + (lane & 15);
    unsigned* o = out + (size_t)t * 8;
#pragma unroll
    for (int j = 0; j < 8; ++j) {
        int k0 = ks * 32 + ((j < 4) ? (hh * 8 + 2 * j) : (16 + hh * 8 + 2 * (j - 4)));
        float a = (k0     < kvalid && n < ncols) ? W[(size_t)k0 * ncols + n]       : 0.0f;
        float b = (k0 + 1 < kvalid && n < ncols) ? W[(size_t)(k0 + 1) * ncols + n] : 0.0f;
        o[j] = pk_bf16(a, b);
    }
}

// ---------------- WMMA GEMM: C[nrows,NOUT] = A[nrows,LDA(f32)] @ B ----------------
// One wave computes a 16-row x (NT*16)-col tile. Packed B is staged once per
// block in LDS (64KB layer1 / 6KB layer2; WGP has 320KB) and read via ds_load.
// A operand (bf16 16x32, ISA 7.12.2): lane L holds row (L&15); VGPR j holds
// K = kb + (j<4?0:16) + (L>>4)*8 + 2*(j&3)  -> two contiguous 32B f32 windows/lane.
template <int LDA, int KVALID, int KSTEPS, int NT, int NOUT>
__global__ void k_gemm_wmma(const float* __restrict__ A, const unsigned* __restrict__ pB,
                            float* __restrict__ C, int nrows) {
    __shared__ __align__(32) unsigned sB[KSTEPS * NT * 32 * 8];

    // cooperative coalesced copy of packed B into LDS (before any wave exits)
    {
        const int total16 = KSTEPS * NT * 32 * 8 / 4;   // uint4 count
        const uint4* g = (const uint4*)pB;
        uint4* s = (uint4*)sB;
        for (int i = threadIdx.x; i < total16; i += blockDim.x) s[i] = g[i];
    }
    __syncthreads();

    int gt   = blockIdx.x * blockDim.x + threadIdx.x;
    int wave = gt >> 5;
    int lane = gt & 31;
    int row0 = wave * 16;
    if (row0 >= nrows) return;                 // wave-uniform: EXEC stays all-ones
    int hh = lane >> 4;
    int ml = lane & 15;
    int mrow = row0 + ml;
    if (mrow >= nrows) mrow = nrows - 1;       // clamp loads, keep EXEC full
    const float* arow = A + (size_t)mrow * LDA;

    v8f acc[NT];
#pragma unroll
    for (int nt = 0; nt < NT; ++nt)
#pragma unroll
        for (int q = 0; q < 8; ++q) acc[nt][q] = 0.0f;

    for (int ks = 0; ks < KSTEPS; ++ks) {
        int kb = ks * 32;
        int c0 = kb + hh * 8;
        int c1 = kb + 16 + hh * 8;
        unsigned au[8];
        if (kb + 32 <= KVALID) {
            const float4* p0 = (const float4*)(arow + c0);
            const float4* p1 = (const float4*)(arow + c1);
            float4 f0 = p0[0], f1 = p0[1], f2 = p1[0], f3 = p1[1];
            if (kb + 64 <= KVALID) __builtin_prefetch(arow + c0 + 32, 0, 3);
            au[0] = pk_bf16(f0.x, f0.y); au[1] = pk_bf16(f0.z, f0.w);
            au[2] = pk_bf16(f1.x, f1.y); au[3] = pk_bf16(f1.z, f1.w);
            au[4] = pk_bf16(f2.x, f2.y); au[5] = pk_bf16(f2.z, f2.w);
            au[6] = pk_bf16(f3.x, f3.y); au[7] = pk_bf16(f3.z, f3.w);
        } else {                                // K tail (once per matrix): zero-pad
#pragma unroll
            for (int j = 0; j < 8; ++j) {
                int cb = (j < 4) ? (c0 + 2 * j) : (c1 + 2 * (j - 4));
                float a = (cb     < KVALID) ? arow[cb]     : 0.0f;
                float b = (cb + 1 < KVALID) ? arow[cb + 1] : 0.0f;
                au[j] = pk_bf16(a, b);
            }
        }
        v8u auv;
#pragma unroll
        for (int j = 0; j < 8; ++j) auv[j] = au[j];
        v16bf aV = __builtin_bit_cast(v16bf, auv);

#pragma unroll
        for (int nt = 0; nt < NT; ++nt) {
            const v8u* bp = (const v8u*)(sB + ((ks * NT + nt) * 32 + lane) * 8);
            v16bf bV = __builtin_bit_cast(v16bf, *bp);     // ds_load_b128 x2
            acc[nt] = __builtin_amdgcn_wmma_f32_16x16x32_bf16(
                false, aV, false, bV, (short)0, acc[nt], false, false);
        }
    }

    // D layout: lane gives col n = (lane&15); VGPR r -> row r + (lane>=16 ? 8 : 0).
#pragma unroll
    for (int nt = 0; nt < NT; ++nt) {
        int ncol = nt * 16 + ml;
        if (ncol >= NOUT) continue;
#pragma unroll
        for (int r = 0; r < 8; ++r) {
            int m = row0 + r + hh * 8;
            if (m < nrows) C[(size_t)m * NOUT + ncol] = acc[nt][r];
        }
    }
}

// ---------------- edge scatter (L2-resident atomics) ----------------
// 16 lanes per edge, float4 gather per lane (64 feats).
__global__ void k_scatter64(const float* __restrict__ h, const int* __restrict__ src,
                            const int* __restrict__ dst, const float* __restrict__ dinv,
                            float* __restrict__ agg, int e) {
    int t  = blockIdx.x * blockDim.x + threadIdx.x;
    int ei = t >> 4;
    if (ei >= e) return;
    int part = t & 15;
    int s = src[ei], d = dst[ei];
    float w = dinv[s] * dinv[d];
    float4 v = ((const float4*)(h + (size_t)s * 64))[part];
    float* o = agg + (size_t)d * 64 + part * 4;
    fadd_atomic(o + 0, w * v.x);
    fadd_atomic(o + 1, w * v.y);
    fadd_atomic(o + 2, w * v.z);
    fadd_atomic(o + 3, w * v.w);
}
// 8 lanes per edge, 5 feats each (40 feats).
__global__ void k_scatter40(const float* __restrict__ h, const int* __restrict__ src,
                            const int* __restrict__ dst, const float* __restrict__ dinv,
                            float* __restrict__ agg, int e) {
    int t  = blockIdx.x * blockDim.x + threadIdx.x;
    int ei = t >> 3;
    if (ei >= e) return;
    int part = t & 7;
    int s = src[ei], d = dst[ei];
    float w = dinv[s] * dinv[d];
    const float* hr = h + (size_t)s * 40 + part * 5;
    float* o = agg + (size_t)d * 40 + part * 5;
#pragma unroll
    for (int i = 0; i < 5; ++i) fadd_atomic(o + i, w * hr[i]);
}

// ---------------- finalize ----------------
// h1 = relu(agg1 + dinv^2 * t1 + b1)   (in place into agg1)
__global__ void k_fin1(float* __restrict__ agg, const float* __restrict__ t1,
                       const float* __restrict__ dinv, const float* __restrict__ b1,
                       int total) {
    int i = blockIdx.x * blockDim.x + threadIdx.x;
    if (i >= total) return;
    int row = i >> 6;                  // /64
    int col = i & 63;
    float di = dinv[row];
    float v = agg[i] + di * di * t1[i] + b1[col];
    agg[i] = v > 0.0f ? v : 0.0f;
}
// out = log_softmax(relu(agg2 + dinv^2 * t2 + b2))
__global__ void k_fin2_lsm(const float* __restrict__ agg, const float* __restrict__ t2,
                           const float* __restrict__ dinv, const float* __restrict__ b2,
                           float* __restrict__ out, int n) {
    int i = blockIdx.x * blockDim.x + threadIdx.x;
    if (i >= n) return;
    float d2 = dinv[i] * dinv[i];
    float v[C_OUT];
    float m = -1e30f;
#pragma unroll
    for (int c = 0; c < C_OUT; ++c) {
        float t = agg[(size_t)i * C_OUT + c] + d2 * t2[(size_t)i * C_OUT + c] + b2[c];
        t = t > 0.0f ? t : 0.0f;
        v[c] = t;
        m = fmaxf(m, t);
    }
    float s = 0.0f;
#pragma unroll
    for (int c = 0; c < C_OUT; ++c) s += expf(v[c] - m);
    float ls = m + logf(s);
#pragma unroll
    for (int c = 0; c < C_OUT; ++c) out[(size_t)i * C_OUT + c] = v[c] - ls;
}

extern "C" void kernel_launch(void* const* d_in, const int* in_sizes, int n_in,
                              void* d_out, int out_size, void* d_ws, size_t ws_size,
                              hipStream_t stream) {
    const float* x  = (const float*)d_in[0];
    const int*   ei = (const int*)d_in[1];        // integer inputs -> int32 per harness
    const float* W1 = (const float*)d_in[2];
    const float* b1 = (const float*)d_in[3];
    const float* W2 = (const float*)d_in[4];
    const float* b2 = (const float*)d_in[5];

    const int n = in_sizes[0] / F_IN;             // 100000
    const int e = in_sizes[1] / 2;                // 1600000
    const int* src = ei;
    const int* dst = ei + e;

    // workspace carve-out (256B aligned)
    char*  ws  = (char*)d_ws;
    size_t off = 0;
    auto take = [&](size_t bytes) -> char* {
        char* p = ws + off;
        off += (bytes + 255) & ~(size_t)255;
        return p;
    };
    float*    dinv = (float*)take((size_t)n * 4);            // deg -> dinv (in place)
    float*    t1   = (float*)take((size_t)n * H_HID * 4);    // x @ W1
    float*    agg1 = (float*)take((size_t)n * H_HID * 4);    // agg, then h1
    float*    t2   = (float*)take((size_t)n * C_OUT * 4);    // h1 @ W2
    float*    agg2 = (float*)take((size_t)n * C_OUT * 4);
    unsigned* pW1  = (unsigned*)take((size_t)16 * 4 * 32 * 8 * 4);  // K=512 pad, 64 cols
    unsigned* pW2  = (unsigned*)take((size_t)2 * 3 * 32 * 8 * 4);   // K=64, 48-col pad
    (void)ws_size; (void)n_in; (void)out_size;

    const int B = 256;
    // degree / norm
    k_init_deg<<<(n + B - 1) / B, B, 0, stream>>>(dinv, n);
    k_count_deg<<<(e + B - 1) / B, B, 0, stream>>>(dinv, dst, e);
    k_dinv<<<(n + B - 1) / B, B, 0, stream>>>(dinv, n);
    // weight pre-pack into WMMA B-operand layout
    k_pack_b<<<(16 * 4 * 32 + B - 1) / B, B, 0, stream>>>(W1, pW1, F_IN, H_HID, 16, 4);
    k_pack_b<<<(2 * 3 * 32 + B - 1) / B, B, 0, stream>>>(W2, pW2, H_HID, C_OUT, 2, 3);

    // layer 1
    long threads_g = (long)((n + 15) / 16) * 32;
    k_gemm_wmma<F_IN, F_IN, 16, 4, H_HID>
        <<<(unsigned)((threads_g + 255) / 256), 256, 0, stream>>>(x, pW1, t1, n);
    k_zero<<<(n * H_HID + B - 1) / B, B, 0, stream>>>(agg1, n * H_HID);
    long st1 = (long)e * 16;
    k_scatter64<<<(unsigned)((st1 + B - 1) / B), B, 0, stream>>>(t1, src, dst, dinv, agg1, e);
    k_fin1<<<(n * H_HID + B - 1) / B, B, 0, stream>>>(agg1, t1, dinv, b1, n * H_HID);

    // layer 2
    k_gemm_wmma<H_HID, H_HID, 2, 3, C_OUT>
        <<<(unsigned)((threads_g + 255) / 256), 256, 0, stream>>>(agg1, pW2, t2, n);
    k_zero<<<(n * C_OUT + B - 1) / B, B, 0, stream>>>(agg2, n * C_OUT);
    long st2 = (long)e * 8;
    k_scatter40<<<(unsigned)((st2 + B - 1) / B), B, 0, stream>>>(t2, src, dst, dinv, agg2, e);
    k_fin2_lsm<<<(n + B - 1) / B, B, 0, stream>>>(agg2, t2, dinv, b2, (float*)d_out, n);
}
